// SD_attn_21096879358222
// MI455X (gfx1250) — matrix operand order repaired and checked by reading, not compiled, not run on hardware
//
#include <hip/hip_runtime.h>
#include <hip/hip_bf16.h>

typedef _Float16 h16;
typedef __attribute__((ext_vector_type(16))) _Float16 v16h;
typedef __attribute__((ext_vector_type(8)))  _Float16 v8h;
typedef __attribute__((ext_vector_type(8)))  float    v8f;
typedef __attribute__((ext_vector_type(4)))  float    v4f;

__device__ inline v16h hcat(v8h lo, v8h hi) {
  union { v16h w; v8h h[2]; } u;
  u.h[0] = lo; u.h[1] = hi;
  return u.w;
}
__device__ inline v8f vzero8() {
  v8f z;
  #pragma unroll
  for (int i = 0; i < 8; ++i) z[i] = 0.f;
  return z;
}
__device__ inline v8f wmma_f16(v16h a, v16h b, v8f c) {
  return __builtin_amdgcn_wmma_f32_16x16x32_f16(false, a, false, b, (short)0, c, false, false);
}

// region label of a token row (mask depends only on the image row; see analysis)
__device__ inline int region_of(int tok, int g, int wr) {
  int th = ((tok >> 3) << 1) + (g >> 1);
  int gh = wr * 16 + th;
  return (gh < 120) ? 0 : ((gh < 124) ? 1 : 2);
}

// ---------------- kernel 0: weight fp32 -> f16 -------------------------------
__global__ __launch_bounds__(256) void k_convert(const float* __restrict__ qkv_w,
                                                 const float* __restrict__ proj_w,
                                                 h16* __restrict__ qkvw16,
                                                 h16* __restrict__ projw16) {
  int i = blockIdx.x * 256 + threadIdx.x;
  const int n1 = 3 * 512 * 512;          // 786432
  if (i < n1) qkvw16[i] = (h16)qkv_w[i];
  int j = i - n1;
  if (j >= 0 && j < 512 * 512) projw16[j] = (h16)proj_w[j];
}

// ---------------- kernel 1: gather + QKV GEMM --------------------------------
#define XPAD 520
__global__ __launch_bounds__(256) void k_qkv(const float* __restrict__ x,
                                             const h16* __restrict__ w16,
                                             const float* __restrict__ bias,
                                             h16* __restrict__ q16,
                                             h16* __restrict__ k16,
                                             h16* __restrict__ vT16) {
  __shared__ h16 X[64 * XPAD];           // 64 tokens x 512 f16 (padded)
  int b  = blockIdx.x;                   // sequence id, 0..1023
  int g  = b & 3, bw = b >> 2;
  int batch = bw >> 7, wr = (bw >> 4) & 7, wc = bw & 15;
  int t = threadIdx.x;

  // ---- stage X: gather with roll(-4,-4) + 16x16 window + 2x2 dilation ----
  {
    int n  = t >> 2;
    int th = ((n >> 3) << 1) + (g >> 1);
    int tw = ((n & 7) << 1) + (g & 1);
    int hh = (wr * 16 + th + 4) & 127;
    int ww = (wc * 16 + tw + 4) & 255;
    const float* src = x + (((size_t)batch * 128 + hh) * 256 + ww) * 512 + (t & 3) * 128;
    h16* dst = &X[n * XPAD + (t & 3) * 128];
    #pragma unroll 4
    for (int i = 0; i < 128; i += 4) {
      v4f f = *(const v4f*)(src + i);
      dst[i + 0] = (h16)f.x; dst[i + 1] = (h16)f.y;
      dst[i + 2] = (h16)f.z; dst[i + 3] = (h16)f.w;
    }
  }
  __syncthreads();

  int wv = t >> 5, lane = t & 31;
  int m = lane & 15, half = lane >> 4;

  // 96 column tiles of 16 (3*512 outputs), 12 per wave; 4 M tiles; K=512 -> 16 steps
  for (int ct = wv * 12; ct < wv * 12 + 12; ++ct) {
    int colg = ct * 16 + m;                       // B-lane column == D-lane column
    const h16* wrow = w16 + (size_t)colg * 512 + half * 16;
    float bcol = bias[colg];
    for (int mt = 0; mt < 4; ++mt) {
      v8f acc = vzero8();
      #pragma unroll 4
      for (int ks = 0; ks < 16; ++ks) {
        const h16* ap = &X[(mt * 16 + m) * XPAD + ks * 32 + half * 16];
        v16h a = hcat(*(const v8h*)ap, *(const v8h*)(ap + 8));
        const h16* bp = wrow + ks * 32;
        v16h bm = hcat(*(const v8h*)bp, *(const v8h*)(bp + 8));
        acc = wmma_f16(a, bm, acc);
      }
      int which = colg >> 9;                      // 0=q 1=k 2=v
      int cc = colg & 511;
      int hh = cc >> 6, d = cc & 63;
      if (which == 2) {                           // v: store transposed, contiguous pack
        v8h pk;
        #pragma unroll
        for (int r = 0; r < 8; ++r) pk[r] = (h16)(acc[r] + bcol);
        size_t idx = (((size_t)b * 8 + hh) * 64 + d) * 64 + mt * 16 + half * 8;
        *(v8h*)&vT16[idx] = pk;
      } else {
        h16* dst = (which == 0) ? q16 : k16;
        size_t base = (((size_t)b * 8 + hh) * 64) * 64 + d;
        #pragma unroll
        for (int r = 0; r < 8; ++r) {
          int tok = mt * 16 + half * 8 + r;
          dst[base + (size_t)tok * 64] = (h16)(acc[r] + bcol);
        }
      }
    }
  }
}

// ---------------- kernel 2: RoPE + attention (one wave per seq,head) ---------
#define QPAD 72
__global__ __launch_bounds__(32) void k_attn(const h16* __restrict__ q16,
                                             const h16* __restrict__ k16,
                                             const h16* __restrict__ vT16,
                                             h16* __restrict__ y16) {
  __shared__ h16 qs[64 * QPAD];
  __shared__ h16 ksh[64 * QPAD];
  __shared__ h16 Ps[64 * QPAD];
  int bh = blockIdx.x;                 // 0..8191
  int b  = bh >> 3, h = bh & 7;
  int lane = threadIdx.x;
  int g = b & 3, bw = b >> 2, wr = (bw >> 4) & 7;

  const h16* qg = q16 + (size_t)bh * 64 * 64;
  const h16* kg = k16 + (size_t)bh * 64 * 64;
  const h16* vg = vT16 + (size_t)bh * 64 * 64;

  // ---- stage q (rope * 0.125) and k (rope) into LDS, vectorized b128 ----
  // 64 tokens x 4 chunks of 16 dims; chunk pairs (0,1)=H-axis rope, (2,3)=W-axis
  const float L2INV = 0.83048202372f;  // log2(10000)/16
  for (int it = 0; it < 8; ++it) {
    int tid = it * 32 + lane;
    int n = tid >> 2, ch = tid & 3;
    int pos = (ch < 2) ? (n >> 3) : (n & 7);
    float sgn = (ch & 1) ? 1.f : -1.f;
    int dbase = ch * 16;
    int pbase = (ch & 1) ? (dbase - 16) : (dbase + 16);
    const v8h* qo = (const v8h*)(qg + n * 64 + dbase);
    const v8h* qp = (const v8h*)(qg + n * 64 + pbase);
    const v8h* ko = (const v8h*)(kg + n * 64 + dbase);
    const v8h* kp = (const v8h*)(kg + n * 64 + pbase);
    v8h q0 = qo[0], q1 = qo[1], p0 = qp[0], p1 = qp[1];
    v8h k0 = ko[0], k1 = ko[1], r0 = kp[0], r1 = kp[1];
    v8h oq0, oq1, ok0, ok1;
    #pragma unroll
    for (int j = 0; j < 16; ++j) {
      float inv = exp2f(-(float)j * L2INV);
      float ang = (float)pos * inv;
      float s, c;
      __sincosf(ang, &s, &c);
      float xq = (j < 8) ? (float)q0[j & 7] : (float)q1[j & 7];
      float pq = (j < 8) ? (float)p0[j & 7] : (float)p1[j & 7];
      float xk = (j < 8) ? (float)k0[j & 7] : (float)k1[j & 7];
      float pk = (j < 8) ? (float)r0[j & 7] : (float)r1[j & 7];
      float vq = 0.125f * (xq * c + sgn * pq * s);
      float vk = xk * c + sgn * pk * s;
      if (j < 8) { oq0[j & 7] = (h16)vq; ok0[j & 7] = (h16)vk; }
      else       { oq1[j & 7] = (h16)vq; ok1[j & 7] = (h16)vk; }
    }
    *(v8h*)&qs[n * QPAD + dbase]      = oq0;
    *(v8h*)&qs[n * QPAD + dbase + 8]  = oq1;
    *(v8h*)&ksh[n * QPAD + dbase]     = ok0;
    *(v8h*)&ksh[n * QPAD + dbase + 8] = ok1;
  }

  int m = lane & 15, half = lane >> 4;

  // ---- S = q k^T (+mask), streaming softmax per 16-row block, P -> LDS ----
  for (int mi = 0; mi < 4; ++mi) {
    v8f S[4];
    for (int nj = 0; nj < 4; ++nj) {
      v8f acc = vzero8();
      #pragma unroll
      for (int ksp = 0; ksp < 2; ++ksp) {
        const h16* ap = &qs[(mi * 16 + m) * QPAD + ksp * 32 + half * 16];
        v16h a = hcat(*(const v8h*)ap, *(const v8h*)(ap + 8));
        const h16* bp = &ksh[(nj * 16 + m) * QPAD + ksp * 32 + half * 16];
        v16h bm = hcat(*(const v8h*)bp, *(const v8h*)(bp + 8));
        acc = wmma_f16(a, bm, acc);
      }
      int regj = region_of(nj * 16 + m, g, wr);
      #pragma unroll
      for (int r = 0; r < 8; ++r) {
        int regi = region_of(mi * 16 + half * 8 + r, g, wr);
        if (regi != regj) acc[r] += -1.0e30f;
      }
      S[nj] = acc;
    }
    // row-wise max: row owned by (half, r), spread across 16 lanes x 4 nj tiles
    float mx[8], sm[8];
    #pragma unroll
    for (int r = 0; r < 8; ++r) {
      float v = fmaxf(fmaxf(S[0][r], S[1][r]), fmaxf(S[2][r], S[3][r]));
      for (int msk = 8; msk >= 1; msk >>= 1)
        v = fmaxf(v, __shfl_xor(v, msk, 32));
      mx[r] = v;
      sm[r] = 0.f;
    }
    for (int nj = 0; nj < 4; ++nj)
      #pragma unroll
      for (int r = 0; r < 8; ++r) {
        float e = __expf(S[nj][r] - mx[r]);
        S[nj][r] = e;
        sm[r] += e;
      }
    #pragma unroll
    for (int r = 0; r < 8; ++r) {
      float v = sm[r];
      for (int msk = 8; msk >= 1; msk >>= 1)
        v += __shfl_xor(v, msk, 32);
      sm[r] = 1.f / v;
    }
    for (int nj = 0; nj < 4; ++nj)
      #pragma unroll
      for (int r = 0; r < 8; ++r) {
        int i = mi * 16 + half * 8 + r;
        Ps[i * QPAD + nj * 16 + m] = (h16)(S[nj][r] * sm[r]);
      }
  }

  // ---- O = P @ V (V stored transposed -> B-layout ready) ----
  for (int mi = 0; mi < 4; ++mi) {
    for (int dj = 0; dj < 4; ++dj) {
      v8f acc = vzero8();
      #pragma unroll
      for (int ksp = 0; ksp < 2; ++ksp) {
        const h16* ap = &Ps[(mi * 16 + m) * QPAD + ksp * 32 + half * 16];
        v16h a = hcat(*(const v8h*)ap, *(const v8h*)(ap + 8));
        const h16* bp = vg + (dj * 16 + m) * 64 + ksp * 32 + half * 16;
        v16h bm = hcat(*(const v8h*)bp, *(const v8h*)(bp + 8));
        acc = wmma_f16(a, bm, acc);
      }
      int d = dj * 16 + m;
      #pragma unroll
      for (int r = 0; r < 8; ++r) {
        int i = mi * 16 + half * 8 + r;
        y16[((size_t)b * 64 + i) * 512 + h * 64 + d] = (h16)acc[r];
      }
    }
  }
}

// ---------------- kernel 3: proj GEMM + inverse-permute scatter --------------
#define YPAD 520
__global__ __launch_bounds__(256) void k_proj(const h16* __restrict__ y16,
                                              const h16* __restrict__ w16,
                                              const float* __restrict__ bias,
                                              float* __restrict__ out) {
  __shared__ h16 Y[64 * YPAD];
  int b = blockIdx.x;
  int t = threadIdx.x;

  // ---- stage Y via CDNA5 async global->LDS DMA (ASYNCcnt path) ----
  {
    int n = t >> 2, qd = t & 3;
    unsigned long long src =
        (unsigned long long)(const void*)(y16 + ((size_t)b * 64 + n) * 512 + qd * 128);
    unsigned int ldsoff =
        (unsigned int)(unsigned long long)(void*)&Y[n * YPAD + qd * 128];
    #pragma unroll
    for (int i = 0; i < 16; ++i) {   // 16 x b128 = 256B per thread
      asm volatile("global_load_async_to_lds_b128 %0, %1, off"
                   :: "v"(ldsoff + i * 16), "v"(src + i * 16)
                   : "memory");
    }
    asm volatile("s_wait_asynccnt 0x0" ::: "memory");
  }
  __syncthreads();

  int wv = t >> 5, lane = t & 31;
  int m = lane & 15, half = lane >> 4;
  int g = b & 3, bw = b >> 2;
  int batch = bw >> 7, wr = (bw >> 4) & 7, wc = bw & 15;

  for (int ct = wv * 4; ct < wv * 4 + 4; ++ct) {      // 32 col tiles, 4/wave
    int colg = ct * 16 + m;
    const h16* wrow = w16 + (size_t)colg * 512 + half * 16;
    float bcol = bias[colg];
    for (int mt = 0; mt < 4; ++mt) {
      v8f acc = vzero8();
      #pragma unroll 4
      for (int ks = 0; ks < 16; ++ks) {
        const h16* ap = &Y[(mt * 16 + m) * YPAD + ks * 32 + half * 16];
        v16h a = hcat(*(const v8h*)ap, *(const v8h*)(ap + 8));
        const h16* bp = wrow + ks * 32;
        v16h bm = hcat(*(const v8h*)bp, *(const v8h*)(bp + 8));
        acc = wmma_f16(a, bm, acc);
      }
      #pragma unroll
      for (int r = 0; r < 8; ++r) {
        int tok = mt * 16 + half * 8 + r;
        int th = ((tok >> 3) << 1) + (g >> 1);
        int tw = ((tok & 7) << 1) + (g & 1);
        int hh = (wr * 16 + th + 4) & 127;
        int ww = (wc * 16 + tw + 4) & 255;
        out[(((size_t)batch * 128 + hh) * 256 + ww) * 512 + colg] = acc[r] + bcol;
      }
    }
  }
}

extern "C" void kernel_launch(void* const* d_in, const int* in_sizes, int n_in,
                              void* d_out, int out_size, void* d_ws, size_t ws_size,
                              hipStream_t stream) {
  const float* x      = (const float*)d_in[0];
  const float* qkv_w  = (const float*)d_in[1];
  const float* qkv_b  = (const float*)d_in[2];
  const float* proj_w = (const float*)d_in[3];
  const float* proj_b = (const float*)d_in[4];

  char* ws = (char*)d_ws;
  const size_t QKVW = 0;
  const size_t PRJW = QKVW + (size_t)3 * 512 * 512 * sizeof(h16);   // 1.5 MB
  const size_t QO   = PRJW + (size_t)512 * 512 * sizeof(h16);       // +0.5 MB
  const size_t KO   = QO   + (size_t)1024 * 8 * 64 * 64 * sizeof(h16); // 64 MB each
  const size_t VO   = KO   + (size_t)1024 * 8 * 64 * 64 * sizeof(h16);
  const size_t YO   = VO   + (size_t)1024 * 8 * 64 * 64 * sizeof(h16);

  h16* qkvw16 = (h16*)(ws + QKVW);
  h16* projw16= (h16*)(ws + PRJW);
  h16* q16    = (h16*)(ws + QO);
  h16* k16    = (h16*)(ws + KO);
  h16* vT16   = (h16*)(ws + VO);
  h16* y16    = (h16*)(ws + YO);

  k_convert<<<4096, 256, 0, stream>>>(qkv_w, proj_w, qkvw16, projw16);
  k_qkv<<<1024, 256, 0, stream>>>(x, qkvw16, qkv_b, q16, k16, vT16);
  k_attn<<<8192, 32, 0, stream>>>(q16, k16, vT16, y16);
  k_proj<<<1024, 256, 0, stream>>>(y16, projw16, proj_b, (float*)d_out);
}
